// LSTM_17746804867561
// MI455X (gfx1250) — matrix operand order repaired and checked by reading, not compiled, not run on hardware
//
#include <hip/hip_runtime.h>

// ---------------------------------------------------------------------------
// LSTM  (B=32, S=512, D=1024) for MI455X / gfx1250, wave32 + WMMA f16.
//
//  Phase 1 (parallel):  x_proj[s][b][e] = x[b][s][:] . U[:,e] + bias[e]
//                       -> 137 GFLOP GEMM on v_wmma_f32_16x16x32_f16
//  Phase 2 (sequential, 512 steps; latency-critical):
//      z = x_proj[s] + h_prev @ W   (W = 8 MB f16, L2-resident across steps)
//      gates -> c_t, h_t
//
//  Optimizations this round:
//   * explicit double-buffered fragment loads (overlap b128 loads with WMMA;
//     kills the s_wait_loadcnt 0x0 stall before every v_wmma)
//   * 4-way K-split in the recurrent GEMM: 512 waves instead of 128, 8
//     K-iterations per wave -> ~4x shorter sequential dependency chain.
//     Partial sums (fp32) are reduced inside the pointwise gate kernel.
// ---------------------------------------------------------------------------

typedef _Float16 half_t;
typedef __attribute__((ext_vector_type(16))) _Float16 v16h;
typedef __attribute__((ext_vector_type(8)))  _Float16 v8h;
typedef __attribute__((ext_vector_type(8)))  float    v8f;

#define D_IN   1024
#define D4     4096
#define BATCH  32
#define SEQ    512
#define KTILES 32            // 1024 / 32
#define KSPLIT 4
#define KSPAN  (KTILES / KSPLIT)   // 8

union Frag16 { v16h v; v8h h[2]; };

static __device__ __forceinline__ float sigf(float x) {
    return 1.0f / (1.0f + __expf(-x));
}
static __device__ __forceinline__ float tanh_fast(float x) {
    float e = __expf(-2.0f * x);
    return (1.0f - e) / (1.0f + e);
}

// ---------------------------------------------------------------------------
// Pack a [D,4D] weight (4 concatenated [D,D] gate matrices) into WMMA
// B-fragment tiles. Tile (nt,kt): N = nt*16..+15, K = kt*32..+31.
// element (p*8+h3) of lane's v16h  <->  W[k][n],
//   n = nt*16 + (lane&15),  k = kt*32 + (lane>>4)*8 + p*16 + h3.
// ---------------------------------------------------------------------------
__global__ void pack_weights(const float* __restrict__ m0, const float* __restrict__ m1,
                             const float* __restrict__ m2, const float* __restrict__ m3,
                             half_t* __restrict__ dst) {
    unsigned tid   = blockIdx.x * blockDim.x + threadIdx.x;   // [0, 4096*1024)
    unsigned tile  = tid >> 9;
    unsigned inner = tid & 511;
    unsigned p    = inner >> 8;
    unsigned lane = (inner >> 3) & 31;
    unsigned h3   = inner & 7;
    unsigned nt   = tile >> 5;                                // [0,256)
    unsigned kt   = tile & 31;
    unsigned n = nt * 16 + (lane & 15);
    unsigned k = kt * 32 + (lane >> 4) * 8 + p * 16 + h3;
    const float* src = (n < 1024) ? m0 : (n < 2048) ? m1 : (n < 3072) ? m2 : m3;
    dst[tid] = (half_t)src[(size_t)k * D_IN + (n & 1023)];
}

__global__ void convert_x(const float* __restrict__ x, half_t* __restrict__ xf) {
    size_t i = (size_t)blockIdx.x * blockDim.x + threadIdx.x;
    xf[i] = (half_t)x[i];
}

__global__ void init_state(float* __restrict__ c, half_t* __restrict__ h) {
    unsigned i = blockIdx.x * blockDim.x + threadIdx.x;       // [0, 32768)
    c[i] = 0.0f;
    h[i] = (half_t)0.0f;
}

// --------------------- fragment load helpers -------------------------------
static __device__ __forceinline__ void load_a(Frag16& f, const half_t* arow, unsigned kt) {
    f.h[0] = *(const v8h*)(arow + kt * 32);
    f.h[1] = *(const v8h*)(arow + kt * 32 + 16);
}
static __device__ __forceinline__ void load_b(Frag16& f, const half_t* pack,
                                              unsigned nt, unsigned kt, unsigned lane) {
    const half_t* tb = pack + (((size_t)nt * KTILES + kt) << 9);
    f.h[0] = *(const v8h*)(tb + lane * 8);
    f.h[1] = *(const v8h*)(tb + 256 + lane * 8);
}

// ---------------------------------------------------------------------------
// Phase 1: x_proj = x @ U + bias.  M = S*B = 16384 (row r = s*32+b),
// N = 4096, K = 1024. Wave tile 16x64 (4 accumulators), double-buffered.
// ---------------------------------------------------------------------------
__global__ void __launch_bounds__(256) xproj_gemm(
    const half_t* __restrict__ xf, const half_t* __restrict__ Upack,
    const float* __restrict__ bi, const float* __restrict__ bfv,
    const float* __restrict__ bo, const float* __restrict__ bg,
    float* __restrict__ xproj)
{
    unsigned wid  = (blockIdx.x * blockDim.x + threadIdx.x) >> 5;
    unsigned lane = threadIdx.x & 31;
    unsigned mt   = wid >> 6;                       // [0,1024)
    unsigned nt4  = wid & 63;                       // [0,64)
    unsigned grp  = lane >> 4;
    unsigned rowA = mt * 16 + (lane & 15);          // r = s*32 + b
    unsigned b    = rowA & 31;
    unsigned s    = rowA >> 5;
    const half_t* arow = xf + ((size_t)b * SEQ + s) * D_IN + grp * 8;

    unsigned ncolLane = nt4 * 64 + (lane & 15);
    const float* bias[4] = { bi, bfv, bo, bg };
    v8f acc[4];
    #pragma unroll
    for (int j = 0; j < 4; ++j) {
        unsigned col = ncolLane + j * 16;
        float bv = bias[col >> 10][col & 1023];
        #pragma unroll
        for (int i = 0; i < 8; ++i) acc[j][i] = bv;
    }

    Frag16 a_cur, b_cur[4];
    load_a(a_cur, arow, 0);
    #pragma unroll
    for (int j = 0; j < 4; ++j) load_b(b_cur[j], Upack, nt4 * 4 + j, 0, lane);

    #pragma unroll 2
    for (unsigned kt = 0; kt < KTILES; ++kt) {
        unsigned ktn = (kt + 1 < KTILES) ? kt + 1 : kt;   // clamped prefetch
        Frag16 a_nxt, b_nxt[4];
        load_a(a_nxt, arow, ktn);
        #pragma unroll
        for (int j = 0; j < 4; ++j) load_b(b_nxt[j], Upack, nt4 * 4 + j, ktn, lane);
        #pragma unroll
        for (int j = 0; j < 4; ++j)
            acc[j] = __builtin_amdgcn_wmma_f32_16x16x32_f16(
                false, a_cur.v, false, b_cur[j].v, (short)0, acc[j], false, false);
        a_cur = a_nxt;
        #pragma unroll
        for (int j = 0; j < 4; ++j) b_cur[j] = b_nxt[j];
    }

    unsigned row0 = mt * 16 + grp * 8;
    #pragma unroll
    for (int j = 0; j < 4; ++j) {
        unsigned col = ncolLane + j * 16;
        #pragma unroll
        for (int i = 0; i < 8; ++i)
            xproj[(size_t)(row0 + i) * D4 + col] = acc[j][i];
    }
}

// ---------------------------------------------------------------------------
// Recurrent step GEMM, K-split 4-way: zpart[ks] = h[:, ksK..] @ W[ksK.., :].
// 512 waves (4 ksplit * 2 mt * 64 nt4), each 8 K-iterations, double-buffered.
// ---------------------------------------------------------------------------
__global__ void __launch_bounds__(256) step_gemm(
    const half_t* __restrict__ hf, const half_t* __restrict__ Wpack,
    float* __restrict__ zpart)
{
    unsigned wid  = (blockIdx.x * blockDim.x + threadIdx.x) >> 5;  // [0,512)
    unsigned lane = threadIdx.x & 31;
    unsigned ks   = wid >> 7;                       // [0,4) K-split slice
    unsigned rem  = wid & 127;
    unsigned mt   = rem >> 6;                       // 0 or 1
    unsigned nt4  = rem & 63;
    unsigned grp  = lane >> 4;
    unsigned rowA = mt * 16 + (lane & 15);          // batch index
    const half_t* arow = hf + (size_t)rowA * D_IN + grp * 8;
    unsigned kt0 = ks * KSPAN;

    v8f acc[4];
    #pragma unroll
    for (int j = 0; j < 4; ++j)
        #pragma unroll
        for (int i = 0; i < 8; ++i) acc[j][i] = 0.0f;

    Frag16 a_cur, b_cur[4];
    load_a(a_cur, arow, kt0);
    #pragma unroll
    for (int j = 0; j < 4; ++j) load_b(b_cur[j], Wpack, nt4 * 4 + j, kt0, lane);

    #pragma unroll
    for (unsigned t = 0; t < KSPAN; ++t) {
        unsigned ktn = kt0 + ((t + 1 < KSPAN) ? t + 1 : t);   // clamped prefetch
        Frag16 a_nxt, b_nxt[4];
        load_a(a_nxt, arow, ktn);
        #pragma unroll
        for (int j = 0; j < 4; ++j) load_b(b_nxt[j], Wpack, nt4 * 4 + j, ktn, lane);
        #pragma unroll
        for (int j = 0; j < 4; ++j)
            acc[j] = __builtin_amdgcn_wmma_f32_16x16x32_f16(
                false, a_cur.v, false, b_cur[j].v, (short)0, acc[j], false, false);
        a_cur = a_nxt;
        #pragma unroll
        for (int j = 0; j < 4; ++j) b_cur[j] = b_nxt[j];
    }

    // zpart layout: [KSPLIT][BATCH][D4]
    unsigned row0 = mt * 16 + grp * 8;
    unsigned ncol = nt4 * 64 + (lane & 15);
    float* zp = zpart + (size_t)ks * BATCH * D4;
    #pragma unroll
    for (int j = 0; j < 4; ++j) {
        unsigned col = ncol + j * 16;
        #pragma unroll
        for (int i = 0; i < 8; ++i)
            zp[(size_t)(row0 + i) * D4 + col] = acc[j][i];
    }
}

// ---------------------------------------------------------------------------
// Pointwise gates + K-split reduction + x_proj add.
// NOTE nonstandard reference math: c_t = sigmoid(f*c + i*g).
// ---------------------------------------------------------------------------
__global__ void step_pointwise(const float* __restrict__ zpart,
                               const float* __restrict__ xproj_s,
                               float* __restrict__ c,
                               half_t* __restrict__ hf, float* __restrict__ out, int s)
{
    unsigned tid = blockIdx.x * blockDim.x + threadIdx.x;   // [0, 32768)
    unsigned b = tid >> 10;
    unsigned d = tid & 1023;

    float zg4[4];
    #pragma unroll
    for (int g = 0; g < 4; ++g) {
        size_t idx = (size_t)b * D4 + g * 1024 + d;
        float v = xproj_s[idx];
        #pragma unroll
        for (int ks = 0; ks < KSPLIT; ++ks)
            v += zpart[(size_t)ks * BATCH * D4 + idx];
        zg4[g] = v;
    }
    float i_t = sigf(zg4[0]);
    float f_t = sigf(zg4[1]);
    float o_t = sigf(zg4[2]);
    float g_t = tanh_fast(zg4[3]);
    float c_t = sigf(f_t * c[tid] + i_t * g_t);
    float h_t = tanh_fast(c_t) * o_t;
    c[tid]  = c_t;
    hf[tid] = (half_t)h_t;
    out[((size_t)b * SEQ + s) * D_IN + d] = h_t;
}

// ---------------------------------------------------------------------------
extern "C" void kernel_launch(void* const* d_in, const int* in_sizes, int n_in,
                              void* d_out, int out_size, void* d_ws, size_t ws_size,
                              hipStream_t stream)
{
    // setup_inputs() order: x, Ui, Wi, Uf, Wf, Uo, Wo, Ug, Wg, bi, bf, bo, bg
    const float* x  = (const float*)d_in[0];
    const float* Ui = (const float*)d_in[1];
    const float* Wi = (const float*)d_in[2];
    const float* Uf = (const float*)d_in[3];
    const float* Wf = (const float*)d_in[4];
    const float* Uo = (const float*)d_in[5];
    const float* Wo = (const float*)d_in[6];
    const float* Ug = (const float*)d_in[7];
    const float* Wg = (const float*)d_in[8];
    const float* bi = (const float*)d_in[9];
    const float* bf = (const float*)d_in[10];
    const float* bo = (const float*)d_in[11];
    const float* bg = (const float*)d_in[12];
    float* out = (float*)d_out;

    char* ws = (char*)d_ws;
    half_t* xf16  = (half_t*)(ws);                        //  32 MiB
    half_t* Upack = (half_t*)(ws + 33554432);             //   8 MiB
    half_t* Wpack = (half_t*)(ws + 41943040);             //   8 MiB
    float*  xproj = (float*) (ws + 50331648);             // 256 MiB  [S][B][4D]
    float*  zpart = (float*) (ws + 318767104);            //   2 MiB  [4][B][4D]
    half_t* hf    = (half_t*)(ws + 320864256);            //  64 KiB  [B][D] f16
    float*  cst   = (float*) (ws + 320929792);            // 128 KiB  [B][D]

    convert_x   <<<65536, 256, 0, stream>>>(x, xf16);
    pack_weights<<<16384, 256, 0, stream>>>(Ui, Uf, Uo, Ug, Upack);
    pack_weights<<<16384, 256, 0, stream>>>(Wi, Wf, Wo, Wg, Wpack);
    init_state  <<<  128, 256, 0, stream>>>(cst, hf);

    // 1024 M-tiles * 64 N-tiles = 65536 waves -> 8192 blocks of 8 waves
    xproj_gemm<<<8192, 256, 0, stream>>>(xf16, Upack, bi, bf, bo, bg, xproj);

    for (int s = 0; s < SEQ; ++s) {
        step_gemm     <<< 64, 256, 0, stream>>>(hf, Wpack, zpart);
        step_pointwise<<<128, 256, 0, stream>>>(zpart,
                                                xproj + (size_t)s * BATCH * D4,
                                                cst, hf, out, s);
    }
}
